// Attention_80341658239681
// MI455X (gfx1250) — matrix operand order repaired and checked by reading, not compiled, not run on hardware
//
#include <hip/hip_runtime.h>

// ---------------------------------------------------------------------------
// Fused attention-without-softmax for MI455X (gfx1250, wave32, WMMA).
//   out = ( [x Wq^T + bq] * ( [x Wk^T + bk]^T [x Wv^T + bv] ) / 8 ) Wp^T + bp
// using the exact identity (qk^T)v = q(k^T v)  (no softmax in the reference),
// which removes the [B,H,T,T] intermediate (805 MB) and 48 GFLOP.
// Matmuls: v_wmma_f32_16x16x32_bf16. GEMM tiles staged into double-buffered
// LDS with GLOBAL_LOAD_ASYNC_TO_LDS_B128 (ASYNCcnt + s_wait_asynccnt).
// ---------------------------------------------------------------------------

#define BB 16
#define TT 1024
#define CC 768
#define HH 12
#define DD 64

typedef __bf16 v16bf __attribute__((ext_vector_type(16)));
typedef __bf16 v8bf  __attribute__((ext_vector_type(8)));
typedef float  v8f   __attribute__((ext_vector_type(8)));

#define WMMA_BF16(a, b, c) \
  __builtin_amdgcn_wmma_f32_16x16x32_bf16(false, (a), false, (b), (short)0, (c), false, false)

static __device__ inline v16bf cat8(v8bf lo, v8bf hi) {
  return __builtin_shufflevector(lo, hi, 0,1,2,3,4,5,6,7,8,9,10,11,12,13,14,15);
}

static __device__ inline v8bf cvt8(const float4 f0, const float4 f1) {
  v8bf r;
  r[0] = (__bf16)f0.x; r[1] = (__bf16)f0.y; r[2] = (__bf16)f0.z; r[3] = (__bf16)f0.w;
  r[4] = (__bf16)f1.x; r[5] = (__bf16)f1.y; r[6] = (__bf16)f1.z; r[7] = (__bf16)f1.w;
  return r;
}

// Async global->LDS copy, 16 bytes per lane; tracked by ASYNCcnt.
// (CDNA5 ISA: VDST = per-lane LDS byte address VGPR,
//  VADDR = per-lane 64-bit global address, SADDR = off.)
static __device__ inline void async_g2l_b128(unsigned lds_byte_addr, const void* gptr) {
  asm volatile("global_load_async_to_lds_b128 %0, %1, off"
               :: "v"(lds_byte_addr), "v"(gptr)
               : "memory");
}

static __device__ inline void wait_async0() {
#if __has_builtin(__builtin_amdgcn_s_wait_asynccnt)
  __builtin_amdgcn_s_wait_asynccnt(0);
#else
  asm volatile("s_wait_asynccnt 0x0" ::: "memory");
#endif
}

// Low 32 bits of a flat pointer into the LDS aperture == LDS byte address.
static __device__ inline unsigned lds_addr(const void* p) {
  return (unsigned)(unsigned long long)p;
}

// ---------------------------------------------------------------------------
// One-shot f32 -> bf16 conversion (x and the four weight matrices).
// ---------------------------------------------------------------------------
__global__ __launch_bounds__(256) void cvt_bf16_kernel(
    const float* __restrict__ src, __bf16* __restrict__ dst, int n) {
  const int i = (blockIdx.x * 256 + threadIdx.x) * 8;
  if (i < n) {
    const float4 f0 = ((const float4*)(src + i))[0];
    const float4 f1 = ((const float4*)(src + i))[1];
    *(v8bf*)(dst + i) = cvt8(f0, f1);
  }
}

// ---------------------------------------------------------------------------
// GEMM:  Out[M,N] = A[M,K] @ W[N,K]^T + bias[N]     (M=16384, N=K=768, bf16)
//   256 threads = 8 waves; block tile 128x128; wave tile 32x64 (2x4 WMMA).
//   K-step 32. A/W tiles live in double-buffered LDS (rows padded to 40 bf16
//   = 80 B -> conflict-free b128 fragment reads), filled by async copies:
//     wait_asynccnt(0) ; barrier ; issue copies for step i+1 into buf (i+1)&1
//     ; 8x WMMA on buf i&1.
// ---------------------------------------------------------------------------
template <bool OUT_BF16>
__global__ __launch_bounds__(256) void gemm_wmma_kernel(
    const __bf16* __restrict__ A, const __bf16* __restrict__ W,
    const float* __restrict__ bias, void* __restrict__ Out) {
  constexpr int Kd  = CC;
  constexpr int Nd  = CC;
  constexpr int LDT = 40;               // padded row length (bf16 elems)
  constexpr int BUFE = 128 * LDT;       // elems per buffer
  constexpr unsigned BUFBYTES = BUFE * 2;

  __shared__ __bf16 As[2 * BUFE];
  __shared__ __bf16 Bs[2 * BUFE];

  const int tid  = threadIdx.x;
  const int lane = tid & 31;
  const int wid  = tid >> 5;
  const int wm   = wid & 3;   // 32-row strip
  const int wn   = wid >> 2;  // 64-col strip
  const int ln   = lane & 15;
  const bool hiL = lane >= 16;

  const int m_blk = blockIdx.x * 128;
  const int n_blk = blockIdx.y * 128;

  // loader mapping: 2 threads per tile row, 16 K-elems (2 x b128) each
  const int lrow = tid >> 1;
  const int lk   = (tid & 1) * 16;

  const __bf16* gA = A + (size_t)(m_blk + lrow) * Kd + lk;
  const __bf16* gW = W + (size_t)(n_blk + lrow) * Kd + lk;
  const unsigned ldsA = lds_addr(&As[lrow * LDT + lk]);
  const unsigned ldsB = lds_addr(&Bs[lrow * LDT + lk]);

  // prologue: stage K-step 0 into buffer 0
  async_g2l_b128(ldsA,      gA);
  async_g2l_b128(ldsA + 16, gA + 8);
  async_g2l_b128(ldsB,      gW);
  async_g2l_b128(ldsB + 16, gW + 8);

  v8f acc[2][4];
#pragma unroll
  for (int s = 0; s < 2; ++s)
#pragma unroll
    for (int j = 0; j < 4; ++j)
      acc[s][j] = (v8f){0.f, 0.f, 0.f, 0.f, 0.f, 0.f, 0.f, 0.f};

  constexpr int nIter = Kd / 32;  // 24
  for (int i = 0; i < nIter; ++i) {
    wait_async0();     // my copies for buffer i&1 have landed
    __syncthreads();   // everyone's copies landed, everyone's prior reads done

    if (i + 1 < nIter) {  // pipeline: fill the other buffer while we compute
      const int ks = (i + 1) * 32;
      const unsigned bo = (unsigned)((i + 1) & 1) * BUFBYTES;
      async_g2l_b128(ldsA + bo,      gA + ks);
      async_g2l_b128(ldsA + bo + 16, gA + ks + 8);
      async_g2l_b128(ldsB + bo,      gW + ks);
      async_g2l_b128(ldsB + bo + 16, gW + ks + 8);
    }

    const int bufe = (i & 1) * BUFE;

    // A frag (ISA 7.12.2): lanes 0-15 row M=ln hold K {0..7,16..23};
    //                      lanes 16-31 hold K {8..15,24..31}
    v16bf af[2];
    const int klo = hiL ? 8 : 0;
#pragma unroll
    for (int s = 0; s < 2; ++s) {
      const __bf16* p = &As[bufe + (wm * 32 + s * 16 + ln) * LDT + klo];
      af[s] = cat8(*(const v8bf*)p, *(const v8bf*)(p + 16));
    }
    // B frag: lanes 0-15 col N=ln hold K {0..7,16..23}; lanes 16-31 hold K {8..15,24..31}
    v16bf bfw[4];
    const int kst = hiL ? 8 : 0;
#pragma unroll
    for (int j = 0; j < 4; ++j) {
      const __bf16* p = &Bs[bufe + (wn * 64 + j * 16 + ln) * LDT + kst];
      bfw[j] = cat8(*(const v8bf*)p, *(const v8bf*)(p + 16));
    }
#pragma unroll
    for (int s = 0; s < 2; ++s)
#pragma unroll
      for (int j = 0; j < 4; ++j)
        acc[s][j] = WMMA_BF16(af[s], bfw[j], acc[s][j]);
  }

  // epilogue: bias + store (D layout: lanes 0-15 rows M=r, lanes 16-31 M=8+r)
#pragma unroll
  for (int j = 0; j < 4; ++j) {
    const int n = n_blk + wn * 64 + j * 16 + ln;
    const float bv_ = bias[n];
#pragma unroll
    for (int s = 0; s < 2; ++s) {
      const int mbase = m_blk + wm * 32 + s * 16 + (hiL ? 8 : 0);
#pragma unroll
      for (int r = 0; r < 8; ++r) {
        const float val = acc[s][j][r] + bv_;
        const size_t off = (size_t)(mbase + r) * Nd + n;
        if constexpr (OUT_BF16)
          ((__bf16*)Out)[off] = (__bf16)val;
        else
          ((float*)Out)[off] = val;
      }
    }
  }
}

// ---------------------------------------------------------------------------
// P[b,h] = (v^T k) * (1/sqrt(D))   -> P = M^T, 64x64 bf16, K-reduce over T.
//   192 blocks (one per b,h), 128 threads = 4 waves; each wave owns a 16-row
//   strip of P. k/v 32x64 tiles are transposed into padded LDS so both
//   fragment reads are contiguous b128s.
// ---------------------------------------------------------------------------
__global__ __launch_bounds__(128) void kv_wmma_kernel(
    const __bf16* __restrict__ kbf, const __bf16* __restrict__ vbf,
    __bf16* __restrict__ P) {
  constexpr int LDT = 40;
  __shared__ __bf16 kT[64 * LDT];
  __shared__ __bf16 vT[64 * LDT];

  const int bh = blockIdx.x;
  const int b  = bh / HH;
  const int h  = bh - b * HH;
  const size_t base = (size_t)b * TT * CC + (size_t)h * DD;

  const int tid  = threadIdx.x;
  const int lane = tid & 31;
  const int wid  = tid >> 5;
  const int ln   = lane & 15;
  const bool hiL = lane >= 16;

  const int trow = tid >> 2;        // 0..31 : t within tile
  const int dgrp = (tid & 3) * 16;  // 0,16,32,48 : d group

  v8f acc[4];
#pragma unroll
  for (int j = 0; j < 4; ++j)
    acc[j] = (v8f){0.f, 0.f, 0.f, 0.f, 0.f, 0.f, 0.f, 0.f};

  for (int ts = 0; ts < TT; ts += 32) {
    const __bf16* kp = kbf + base + (size_t)(ts + trow) * CC + dgrp;
    const v8bf k0 = ((const v8bf*)kp)[0];
    const v8bf k1 = ((const v8bf*)kp)[1];
    const __bf16* vp = vbf + base + (size_t)(ts + trow) * CC + dgrp;
    const v8bf v0 = ((const v8bf*)vp)[0];
    const v8bf v1 = ((const v8bf*)vp)[1];

    __syncthreads();
#pragma unroll
    for (int i = 0; i < 8; ++i) {
      kT[(dgrp + i) * LDT + trow]     = k0[i];
      kT[(dgrp + 8 + i) * LDT + trow] = k1[i];
      vT[(dgrp + i) * LDT + trow]     = v0[i];
      vT[(dgrp + 8 + i) * LDT + trow] = v1[i];
    }
    __syncthreads();

    const int klo = hiL ? 8 : 0;
    const __bf16* pa = &vT[(wid * 16 + ln) * LDT + klo];
    const v16bf af = cat8(*(const v8bf*)pa, *(const v8bf*)(pa + 16));
    const int kst = hiL ? 8 : 0;
#pragma unroll
    for (int j = 0; j < 4; ++j) {
      const __bf16* pb = &kT[(j * 16 + ln) * LDT + kst];
      const v16bf bfw = cat8(*(const v8bf*)pb, *(const v8bf*)(pb + 16));
      acc[j] = WMMA_BF16(af, bfw, acc[j]);
    }
  }

  __bf16* Pp = P + (size_t)bh * DD * DD;
#pragma unroll
  for (int j = 0; j < 4; ++j) {
    const int n  = j * 16 + ln;               // d1
    const int mb = wid * 16 + (hiL ? 8 : 0);  // d2 base
#pragma unroll
    for (int r = 0; r < 8; ++r)
      Pp[(size_t)(mb + r) * DD + n] = (__bf16)(acc[j][r] * 0.125f);
  }
}

// ---------------------------------------------------------------------------
// y[b, t, h*64 + d2] = sum_{d1} q[b,t,h,d1] * M[d1,d2]   (M[d1][d2] = P[d2][d1])
//   Grid (T/128, B*H), 128 threads = 4 waves; wave owns 32 t-rows x all 64 d2.
//   K = 64 -> two WMMA K-steps; fragments straight from global (P is tiny &
//   L2-resident, q rows contiguous).
// ---------------------------------------------------------------------------
__global__ __launch_bounds__(128) void qm_wmma_kernel(
    const __bf16* __restrict__ qbf, const __bf16* __restrict__ P,
    __bf16* __restrict__ y) {
  const int bh = blockIdx.y;
  const int b  = bh / HH;
  const int h  = bh - b * HH;

  const int tid  = threadIdx.x;
  const int lane = tid & 31;
  const int wid  = tid >> 5;
  const int ln   = lane & 15;
  const bool hiL = lane >= 16;

  const int t0 = blockIdx.x * 128 + wid * 32;
  const size_t rowbase = (size_t)b * TT * CC + (size_t)h * DD;
  const __bf16* Pp = P + (size_t)bh * DD * DD;

  v8f acc[2][4];
#pragma unroll
  for (int s = 0; s < 2; ++s)
#pragma unroll
    for (int j = 0; j < 4; ++j)
      acc[s][j] = (v8f){0.f, 0.f, 0.f, 0.f, 0.f, 0.f, 0.f, 0.f};

#pragma unroll
  for (int ks = 0; ks < DD; ks += 32) {
    const int klo = ks + (hiL ? 8 : 0);
    v16bf af[2];
#pragma unroll
    for (int s = 0; s < 2; ++s) {
      const __bf16* p = qbf + rowbase + (size_t)(t0 + s * 16 + ln) * CC + klo;
      af[s] = cat8(((const v8bf*)p)[0], *(const v8bf*)(p + 16));
    }
    const int kst = ks + (hiL ? 8 : 0);
#pragma unroll
    for (int j = 0; j < 4; ++j) {
      const __bf16* p = Pp + (size_t)(j * 16 + ln) * DD + kst;
      const v16bf bfw = cat8(((const v8bf*)p)[0], *(const v8bf*)(p + 16));
#pragma unroll
      for (int s = 0; s < 2; ++s)
        acc[s][j] = WMMA_BF16(af[s], bfw, acc[s][j]);
    }
  }

#pragma unroll
  for (int j = 0; j < 4; ++j) {
    const int col = j * 16 + ln;  // d2
#pragma unroll
    for (int s = 0; s < 2; ++s) {
      const int tb = t0 + s * 16 + (hiL ? 8 : 0);
#pragma unroll
      for (int r = 0; r < 8; ++r)
        y[rowbase + (size_t)(tb + r) * CC + col] = (__bf16)acc[s][j][r];
    }
  }
}

// ---------------------------------------------------------------------------
extern "C" void kernel_launch(void* const* d_in, const int* in_sizes, int n_in,
                              void* d_out, int out_size, void* d_ws, size_t ws_size,
                              hipStream_t stream) {
  const float* x  = (const float*)d_in[0];
  const float* Wq = (const float*)d_in[1];
  const float* bq = (const float*)d_in[2];
  const float* Wk = (const float*)d_in[3];
  const float* bk = (const float*)d_in[4];
  const float* Wv = (const float*)d_in[5];
  const float* bv = (const float*)d_in[6];
  const float* Wp = (const float*)d_in[7];
  const float* bp = (const float*)d_in[8];
  float* out = (float*)d_out;

  const int    MC = BB * TT * CC;  // 12,582,912 elems
  const int    WC = CC * CC;       // 589,824 elems
  __bf16* xbf  = (__bf16*)d_ws;
  __bf16* qbf  = xbf + MC;
  __bf16* kbf  = qbf + MC;
  __bf16* vbf  = kbf + MC;
  __bf16* ybf  = vbf + MC;
  __bf16* wqbf = ybf + MC;
  __bf16* wkbf = wqbf + WC;
  __bf16* wvbf = wkbf + WC;
  __bf16* wpbf = wvbf + WC;
  __bf16* Pbf  = wpbf + WC;  // B*H*64*64; total ws ~132 MB

  // one-shot bf16 conversions
  cvt_bf16_kernel<<<MC / (256 * 8), 256, 0, stream>>>(x, xbf, MC);
  cvt_bf16_kernel<<<WC / (256 * 8), 256, 0, stream>>>(Wq, wqbf, WC);
  cvt_bf16_kernel<<<WC / (256 * 8), 256, 0, stream>>>(Wk, wkbf, WC);
  cvt_bf16_kernel<<<WC / (256 * 8), 256, 0, stream>>>(Wv, wvbf, WC);
  cvt_bf16_kernel<<<WC / (256 * 8), 256, 0, stream>>>(Wp, wpbf, WC);

  const dim3 gG(BB * TT / 128, CC / 128);  // (128, 6)

  gemm_wmma_kernel<true><<<gG, 256, 0, stream>>>(xbf, wqbf, bq, qbf);
  gemm_wmma_kernel<true><<<gG, 256, 0, stream>>>(xbf, wkbf, bk, kbf);
  gemm_wmma_kernel<true><<<gG, 256, 0, stream>>>(xbf, wvbf, bv, vbf);
  kv_wmma_kernel<<<BB * HH, 128, 0, stream>>>(kbf, vbf, Pbf);
  qm_wmma_kernel<<<dim3(TT / 128, BB * HH), 128, 0, stream>>>(qbf, Pbf, ybf);
  gemm_wmma_kernel<false><<<gG, 256, 0, stream>>>(ybf, wpbf, bp, out);
}
